// AttentionBlock_69183333204231
// MI455X (gfx1250) — matrix operand order repaired and checked by reading, not compiled, not run on hardware
//
#include <hip/hip_runtime.h>
#include <hip/hip_bf16.h>

// ---------------------------------------------------------------------------
// AttentionBlock for MI455X (gfx1250): GroupNorm -> QKV GEMM -> flash attn
// -> out-proj + residual.  GEMMs use v_wmma_f32_16x16x32_f16 with 32x64 wave
// tiles; shared operand tiles (weights, K, V) are staged in LDS via the
// gfx1250 async-copy path (global_load_async_to_lds_b128 + s_wait_asynccnt)
// so each block loads them once instead of once per wave.
// ---------------------------------------------------------------------------

#define Bn  16
#define Cc  512
#define Ss  1024          // H*W = 32*32
#define NHh 4
#define HSz 128           // head size
#define EPSV 1e-5f

typedef __attribute__((ext_vector_type(16))) _Float16 v16h;
typedef __attribute__((ext_vector_type(8)))  _Float16 v8h;
typedef __attribute__((ext_vector_type(8)))  float    v8f;

// Async DMA of 16B from global to LDS (gfx1250, tracked by ASYNCcnt).
// Generic pointers to __shared__ hold the LDS byte offset in bits[31:0]
// (ISA 10.2), which is exactly what VDST wants.
__device__ __forceinline__ void async_copy16(const _Float16* gp, _Float16* lp) {
    unsigned lo = (unsigned)(unsigned long long)lp;
    asm volatile("global_load_async_to_lds_b128 %0, %1, off"
                 :: "v"(lo), "v"(gp) : "memory");
}
__device__ __forceinline__ void async_wait() {
    asm volatile("s_wait_asynccnt 0" ::: "memory");
}

// A-matrix 16x32 f16 fragment (row-major source, leading dim ld):
// lane 0-15: M=lane, K-halves {0..7, 16..23}; lane 16-31: M=lane-16,
// K-halves {8..15, 24..31}.  Two 16B loads per lane.
__device__ __forceinline__ v16h load_a_frag(const _Float16* base, int ld) {
    const int lane = threadIdx.x & 31;
    const int m  = lane & 15;
    const int hi = lane >> 4;
    const _Float16* p = base + (size_t)m * ld + hi * 8;
    union { v16h f; v8h h[2]; } u;
    u.h[0] = *(const v8h*)(p);
    u.h[1] = *(const v8h*)(p + 16);
    return u.f;
}

// B-matrix 32x16 f16 fragment.  Source given as B^T row-major (rows = N,
// K contiguous, leading dim ld): lane&15 selects N row; lane 0-15 K-halves
// {0..7, 16..23}, lane 16-31 K-halves {8..15, 24..31} (two 16B loads).
__device__ __forceinline__ v16h load_b_frag(const _Float16* base, int ld) {
    const int lane = threadIdx.x & 31;
    const int n  = lane & 15;
    const int kb = (lane >> 4) * 8;
    const _Float16* p = base + (size_t)n * ld + kb;
    union { v16h f; v8h h[2]; } u;
    u.h[0] = *(const v8h*)(p);
    u.h[1] = *(const v8h*)(p + 16);
    return u.f;
}

__device__ __forceinline__ v8f wmma16(v16h a, v16h b, v8f c) {
    return __builtin_amdgcn_wmma_f32_16x16x32_f16(
        /*neg_a=*/false, a, /*neg_b=*/false, b,
        /*c_mod=*/(short)0, c, /*reuse_a=*/false, /*reuse_b=*/false);
}

// ---------------------------------------------------------------------------
// Kernel 0: per-batch mean / rstd over C*S elements (GroupNorm, 1 group)
// ---------------------------------------------------------------------------
__global__ void gn_stats_kernel(const float* __restrict__ x,
                                float* __restrict__ stats) {
    const int b = blockIdx.x;
    const size_t n = (size_t)Cc * Ss;
    const float* xb = x + (size_t)b * n;
    float s0 = 0.f, s1 = 0.f;
    for (size_t i = threadIdx.x; i < n; i += blockDim.x) {
        float v = xb[i];
        s0 += v; s1 += v * v;
    }
    __shared__ float r0[256], r1[256];
    r0[threadIdx.x] = s0; r1[threadIdx.x] = s1;
    __syncthreads();
    for (int off = 128; off > 0; off >>= 1) {
        if ((int)threadIdx.x < off) {
            r0[threadIdx.x] += r0[threadIdx.x + off];
            r1[threadIdx.x] += r1[threadIdx.x + off];
        }
        __syncthreads();
    }
    if (threadIdx.x == 0) {
        float inv_n = 1.0f / (float)n;
        float mean = r0[0] * inv_n;
        float var  = r1[0] * inv_n - mean * mean;
        stats[2 * b]     = mean;
        stats[2 * b + 1] = rsqrtf(var + EPSV);
    }
}

// ---------------------------------------------------------------------------
// Kernel 1: normalize + affine, transpose (B,C,S) -> (B,S,C), cast to f16
// ---------------------------------------------------------------------------
__global__ void gn_apply_kernel(const float* __restrict__ x,
                                const float* __restrict__ gw,
                                const float* __restrict__ gb,
                                const float* __restrict__ stats,
                                _Float16* __restrict__ y) {
    size_t idx = (size_t)blockIdx.x * blockDim.x + threadIdx.x;
    const size_t total = (size_t)Bn * Cc * Ss;
    if (idx >= total) return;
    int b = (int)(idx / ((size_t)Cc * Ss));
    int rem = (int)(idx % ((size_t)Cc * Ss));
    int c = rem / Ss;
    int s = rem % Ss;
    float mean = stats[2 * b], rstd = stats[2 * b + 1];
    float v = (x[idx] - mean) * rstd * gw[c] + gb[c];
    y[(size_t)b * Ss * Cc + (size_t)s * Cc + c] = (_Float16)v;
}

// ---------------------------------------------------------------------------
// Kernel 2: f32 -> f16 cast (for the weight matrices)
// ---------------------------------------------------------------------------
__global__ void cast_f16_kernel(const float* __restrict__ in,
                                _Float16* __restrict__ out, int n) {
    int i = blockIdx.x * blockDim.x + threadIdx.x;
    if (i < n) out[i] = (_Float16)in[i];
}

// ---------------------------------------------------------------------------
// Kernel 3: QKV GEMM.  qkv[s, j] = sum_c y[s,c] * qkv_w[j,c] + qkv_b[j]
// Block = 128 threads (4 waves).  Block tile 128(M) x 64(N); each wave owns
// a 32x64 sub-tile (2 A-frags x 4 B-frags -> 8 WMMA per K step).  The 64x512
// weight tile is async-DMA'd to LDS once and shared by all 4 waves.
// ---------------------------------------------------------------------------
__global__ __launch_bounds__(128)
void qkv_gemm_kernel(const _Float16* __restrict__ y,
                     const _Float16* __restrict__ w,
                     const float* __restrict__ bias,
                     _Float16* __restrict__ qb,
                     _Float16* __restrict__ kb,
                     _Float16* __restrict__ vtb) {
    const int b    = blockIdx.z;
    const int wave = threadIdx.x >> 5;
    const int row0 = blockIdx.y * 128 + wave * 32;
    const int n0   = blockIdx.x * 64;

    // stage the block's 64 x 512 weight tile in LDS (64KB) via async DMA
    __shared__ _Float16 wt[64 * Cc];
    for (int c = threadIdx.x; c < 4096; c += 128) {
        int row  = c >> 6;           // 0..63
        int col  = (c & 63) * 8;     // halves, 16B chunks
        async_copy16(w + (size_t)(n0 + row) * Cc + col, wt + row * Cc + col);
    }
    async_wait();
    __syncthreads();

    const _Float16* A0 = y + (size_t)b * Ss * Cc + (size_t)row0 * Cc;

    v8f acc[2][4] = {{v8f{}, v8f{}, v8f{}, v8f{}},
                     {v8f{}, v8f{}, v8f{}, v8f{}}};
    for (int k0 = 0; k0 < Cc; k0 += 32) {
        v16h a0 = load_a_frag(A0 + k0, Cc);
        v16h a1 = load_a_frag(A0 + (size_t)16 * Cc + k0, Cc);
#pragma unroll
        for (int t = 0; t < 4; t++) {
            v16h bf = load_b_frag(wt + (size_t)(16 * t) * Cc + k0, Cc);
            acc[0][t] = wmma16(a0, bf, acc[0][t]);
            acc[1][t] = wmma16(a1, bf, acc[1][t]);
        }
    }

    const int lane = threadIdx.x & 31;
    const int ln = lane & 15, hi = lane >> 4;
#pragma unroll
    for (int mi = 0; mi < 2; mi++) {
#pragma unroll
        for (int t = 0; t < 4; t++) {
            int col = n0 + 16 * t + ln;           // j in [0, 3C)
            float bv = bias[col];
            int which = col / Cc;                 // 0=Q 1=K 2=V
            int cc = col % Cc;
            int h = cc / HSz, d = cc % HSz;
#pragma unroll
            for (int r = 0; r < 8; r++) {
                int row = row0 + 16 * mi + r + 8 * hi;
                float val = acc[mi][t][r] + bv;
                if (which == 0)
                    qb[(((size_t)b * NHh + h) * Ss + row) * HSz + d] = (_Float16)val;
                else if (which == 1)
                    kb[(((size_t)b * NHh + h) * Ss + row) * HSz + d] = (_Float16)val;
                else
                    vtb[(((size_t)b * NHh + h) * HSz + d) * Ss + row] = (_Float16)val;
            }
        }
    }
}

// ---------------------------------------------------------------------------
// Kernel 4: flash attention.  grid (S/64, NH, B), block 128 (4 waves).
// Each wave owns 16 query rows; KV processed in tiles of 32 with online
// softmax.  K (32x128) and V^T (128x32) tiles are async-DMA'd to LDS once
// per step and shared by all 4 waves.  P is bounced through a per-wave LDS
// tile to convert C-frag layout -> A-frag layout for the P*V WMMA.
// ---------------------------------------------------------------------------
__global__ __launch_bounds__(128)
void attn_kernel(const _Float16* __restrict__ qb,
                 const _Float16* __restrict__ kbm,
                 const _Float16* __restrict__ vtb,
                 _Float16* __restrict__ ob) {
    const int b = blockIdx.z, h = blockIdx.y;
    const int wave = threadIdx.x >> 5;
    const int row0 = blockIdx.x * 64 + wave * 16;
    const _Float16* Q = qb  + ((size_t)(b * NHh + h) * Ss + row0) * HSz;
    const _Float16* K = kbm + (size_t)(b * NHh + h) * Ss * HSz;
    const _Float16* V = vtb + (size_t)(b * NHh + h) * HSz * Ss;  // (HS, S)

    __shared__ _Float16 kt[32 * HSz];      // K tile, rows = kv, 8KB
    __shared__ _Float16 vts[HSz * 32];     // V^T tile, rows = d, 8KB
    __shared__ _Float16 plds[4][16 * 32];  // per-wave P scratch, 4KB
    _Float16* P = plds[wave];

    const int lane = threadIdx.x & 31;
    const int ln = lane & 15, hi = lane >> 4;
    const float scale = 0.08838834764831845f;  // 1/sqrt(128)

    v16h qf[4];
#pragma unroll
    for (int i = 0; i < 4; i++) qf[i] = load_a_frag(Q + 32 * i, HSz);

    v8f Ov[8];
    float mrow[8], lrow[8];
#pragma unroll
    for (int r = 0; r < 8; r++) { Ov[r] = v8f{}; mrow[r] = -1e30f; lrow[r] = 0.f; }

    for (int kv = 0; kv < Ss; kv += 32) {
        // ---- stage K / V^T tiles for this step (shared by all 4 waves) ----
        __syncthreads();   // prior step's LDS reads must be done
        for (int c = threadIdx.x; c < 512; c += 128) {     // K: 32 x 128
            int row = c >> 2, col = (c & 3) * 32;          // 4 x 16B per row? no:
        }
        // K tile: 32 rows x 128 halves = 512 chunks of 8 halves (16B)
        for (int c = threadIdx.x; c < 512; c += 128) {
            int row = c >> 4;            // 0..31
            int col = (c & 15) * 8;      // 0..120
            async_copy16(K + (size_t)(kv + row) * HSz + col, kt + row * HSz + col);
        }
        // V^T tile: 128 rows x 32 halves = 512 chunks of 8 halves (16B)
        for (int c = threadIdx.x; c < 512; c += 128) {
            int row = c >> 2;            // 0..127
            int col = (c & 3) * 8;       // 0..24
            async_copy16(V + (size_t)row * Ss + kv + col, vts + row * 32 + col);
        }
        async_wait();
        __syncthreads();

        // ---- scores S = Q * K^T (16 x 32), scaled ----
        v8f sc[2] = {v8f{}, v8f{}};
#pragma unroll
        for (int j = 0; j < 2; j++) {
#pragma unroll
            for (int kk = 0; kk < 4; kk++) {
                v16h bf = load_b_frag(kt + (size_t)(16 * j) * HSz + 32 * kk, HSz);
                sc[j] = wmma16(qf[kk], bf, sc[j]);
            }
        }
#pragma unroll
        for (int j = 0; j < 2; j++)
#pragma unroll
            for (int r = 0; r < 8; r++) sc[j][r] *= scale;

        // ---- row max across this KV tile (reduce over 16-lane halves) ----
        float mt[8];
#pragma unroll
        for (int r = 0; r < 8; r++) mt[r] = fmaxf(sc[0][r], sc[1][r]);
#pragma unroll
        for (int off = 1; off < 16; off <<= 1)
#pragma unroll
            for (int r = 0; r < 8; r++)
                mt[r] = fmaxf(mt[r], __shfl_xor(mt[r], off, 32));

        float mnew[8], corr[8], psum[8];
#pragma unroll
        for (int r = 0; r < 8; r++) {
            mnew[r] = fmaxf(mrow[r], mt[r]);
            corr[r] = __expf(mrow[r] - mnew[r]);
            psum[r] = 0.f;
        }

        // ---- P = exp(S - mnew); stash to per-wave LDS in C-frag coords ----
#pragma unroll
        for (int j = 0; j < 2; j++)
#pragma unroll
            for (int r = 0; r < 8; r++) {
                float p = __expf(sc[j][r] - mnew[r]);
                psum[r] += p;
                P[(r + 8 * hi) * 32 + 16 * j + ln] = (_Float16)p;
            }
#pragma unroll
        for (int off = 1; off < 16; off <<= 1)
#pragma unroll
            for (int r = 0; r < 8; r++)
                psum[r] += __shfl_xor(psum[r], off, 32);
#pragma unroll
        for (int r = 0; r < 8; r++) {
            lrow[r] = lrow[r] * corr[r] + psum[r];
            mrow[r] = mnew[r];
        }
        // rescale the running output accumulator
#pragma unroll
        for (int nt = 0; nt < 8; nt++)
#pragma unroll
            for (int r = 0; r < 8; r++) Ov[nt][r] *= corr[r];

        // ---- O += P * V ----  (re-load P as an A fragment from LDS)
        v16h pf = load_a_frag(P, 32);
#pragma unroll
        for (int nt = 0; nt < 8; nt++) {
            v16h vf = load_b_frag(vts + (size_t)(16 * nt) * 32, 32);
            Ov[nt] = wmma16(pf, vf, Ov[nt]);
        }
    }

    // ---- epilogue: normalize by l, store o[b, s, c] (c = h*HS + d) ----
    float inv[8];
#pragma unroll
    for (int r = 0; r < 8; r++) inv[r] = 1.0f / lrow[r];
    _Float16* Orow = ob + (size_t)b * Ss * Cc;
#pragma unroll
    for (int nt = 0; nt < 8; nt++)
#pragma unroll
        for (int r = 0; r < 8; r++) {
            int row = row0 + r + 8 * hi;
            int c = h * HSz + 16 * nt + ln;
            Orow[(size_t)row * Cc + c] = (_Float16)(Ov[nt][r] * inv[r]);
        }
}

// ---------------------------------------------------------------------------
// Kernel 5: output projection + bias + residual, transpose back to (B,C,S)
// out[b,c,s] = x[b,c,s] + sum_c' o[b,s,c'] * out_w[c,c'] + out_b[c]
// Same 32x64 wave tiling + async LDS weight staging as the QKV GEMM.
// ---------------------------------------------------------------------------
__global__ __launch_bounds__(128)
void proj_kernel(const _Float16* __restrict__ o,
                 const _Float16* __restrict__ w,
                 const float* __restrict__ bias,
                 const float* __restrict__ x,
                 float* __restrict__ out) {
    const int b    = blockIdx.z;
    const int wave = threadIdx.x >> 5;
    const int row0 = blockIdx.y * 128 + wave * 32;  // s
    const int n0   = blockIdx.x * 64;               // c

    __shared__ _Float16 wt[64 * Cc];
    for (int c = threadIdx.x; c < 4096; c += 128) {
        int row  = c >> 6;
        int col  = (c & 63) * 8;
        async_copy16(w + (size_t)(n0 + row) * Cc + col, wt + row * Cc + col);
    }
    async_wait();
    __syncthreads();

    const _Float16* A0 = o + (size_t)b * Ss * Cc + (size_t)row0 * Cc;

    v8f acc[2][4] = {{v8f{}, v8f{}, v8f{}, v8f{}},
                     {v8f{}, v8f{}, v8f{}, v8f{}}};
    for (int k0 = 0; k0 < Cc; k0 += 32) {
        v16h a0 = load_a_frag(A0 + k0, Cc);
        v16h a1 = load_a_frag(A0 + (size_t)16 * Cc + k0, Cc);
#pragma unroll
        for (int t = 0; t < 4; t++) {
            v16h bf = load_b_frag(wt + (size_t)(16 * t) * Cc + k0, Cc);
            acc[0][t] = wmma16(a0, bf, acc[0][t]);
            acc[1][t] = wmma16(a1, bf, acc[1][t]);
        }
    }

    const int lane = threadIdx.x & 31;
    const int ln = lane & 15, hi = lane >> 4;
#pragma unroll
    for (int mi = 0; mi < 2; mi++) {
#pragma unroll
        for (int t = 0; t < 4; t++) {
            int c = n0 + 16 * t + ln;
            float bv = bias[c];
#pragma unroll
            for (int r = 0; r < 8; r++) {
                int s = row0 + 16 * mi + r + 8 * hi;
                size_t idx = ((size_t)b * Cc + c) * Ss + s;
                out[idx] = x[idx] + acc[mi][t][r] + bv;
            }
        }
    }
}

// ---------------------------------------------------------------------------
// Host launcher
// ---------------------------------------------------------------------------
extern "C" void kernel_launch(void* const* d_in, const int* in_sizes, int n_in,
                              void* d_out, int out_size, void* d_ws, size_t ws_size,
                              hipStream_t stream) {
    const float* x      = (const float*)d_in[0];
    const float* gn_w   = (const float*)d_in[1];
    const float* gn_b   = (const float*)d_in[2];
    const float* qkv_w  = (const float*)d_in[3];
    const float* qkv_b  = (const float*)d_in[4];
    const float* out_w  = (const float*)d_in[5];
    const float* out_b  = (const float*)d_in[6];
    float* out = (float*)d_out;

    // workspace carve-up (f16 intermediates; ~86 MB total)
    char* ws = (char*)d_ws;
    size_t off = 0;
    auto carve = [&](size_t bytes) {
        char* p = ws + off;
        off = (off + bytes + 255) & ~(size_t)255;
        return (void*)p;
    };
    float*    stats   = (float*)   carve(Bn * 2 * sizeof(float));
    _Float16* qkv_w_h = (_Float16*)carve((size_t)3 * Cc * Cc * 2);
    _Float16* out_w_h = (_Float16*)carve((size_t)Cc * Cc * 2);
    _Float16* y_h     = (_Float16*)carve((size_t)Bn * Ss * Cc * 2);
    _Float16* q_h     = (_Float16*)carve((size_t)Bn * Ss * Cc * 2);
    _Float16* k_h     = (_Float16*)carve((size_t)Bn * Ss * Cc * 2);
    _Float16* vt_h    = (_Float16*)carve((size_t)Bn * Ss * Cc * 2);
    _Float16* o_h     = (_Float16*)carve((size_t)Bn * Ss * Cc * 2);
    (void)ws_size; (void)in_sizes; (void)n_in; (void)out_size;

    // 0) GroupNorm statistics
    gn_stats_kernel<<<Bn, 256, 0, stream>>>(x, stats);

    // 1) normalize + transpose + cast
    {
        size_t total = (size_t)Bn * Cc * Ss;
        gn_apply_kernel<<<(unsigned)((total + 255) / 256), 256, 0, stream>>>(
            x, gn_w, gn_b, stats, y_h);
    }

    // 2) cast weights to f16
    {
        int n1 = 3 * Cc * Cc;
        cast_f16_kernel<<<(n1 + 255) / 256, 256, 0, stream>>>(qkv_w, qkv_w_h, n1);
        int n2 = Cc * Cc;
        cast_f16_kernel<<<(n2 + 255) / 256, 256, 0, stream>>>(out_w, out_w_h, n2);
    }

    // 3) QKV projection
    qkv_gemm_kernel<<<dim3(3 * Cc / 64, Ss / 128, Bn), 128, 0, stream>>>(
        y_h, qkv_w_h, qkv_b, q_h, k_h, vt_h);

    // 4) attention
    attn_kernel<<<dim3(Ss / 64, NHh, Bn), 128, 0, stream>>>(q_h, k_h, vt_h, o_h);

    // 5) output projection + residual
    proj_kernel<<<dim3(Cc / 64, Ss / 128, Bn), 128, 0, stream>>>(
        o_h, out_w_h, out_b, x, out);
}
